// QConv1dINT_30983894073637
// MI455X (gfx1250) — matrix operand order repaired and checked by reading, not compiled, not run on hardware
//
#include <hip/hip_runtime.h>

// ---------------------------------------------------------------------------
// QConv1d with int8 fake-quant, lowered to f16 WMMA implicit GEMM (gfx1250).
//   out[b,o,l] = sum_{c,k} q_x[b,c,l+k-1] * ( q_w[o,c,k]*sw[o]*sa[c] ) + bias[o]
//   A (weights, scales folded) : f16, precomputed into d_ws once per launch
//   B (activations)            : exact int8 values held in f16
//   A chunks double-buffered in LDS via GLOBAL_LOAD_ASYNC_TO_LDS (ASYNCcnt).
// ---------------------------------------------------------------------------

typedef _Float16 v8h  __attribute__((ext_vector_type(8)));
typedef _Float16 v16h __attribute__((ext_vector_type(16)));
typedef float    v8f  __attribute__((ext_vector_type(8)));

typedef int v4i __attribute__((vector_size(4 * sizeof(int))));
typedef __attribute__((address_space(1))) v4i* as1_v4i;
typedef __attribute__((address_space(3))) v4i* as3_v4i;

#define C_IN   512
#define C_OUT  512
#define LEN    4096
#define BATCH  32
#define KW     3
#define QEPS   1e-8f

// LDS layout (dynamic): B slab, then double-buffered A chunks
#define B_ROWSTRIDE 1040              // 512*2 bytes + 16 pad  (16B aligned)
#define B_ROWS      132               // 128 + 2 halo (+2 align), rows = l_local
#define B_BYTES     (B_ROWS * B_ROWSTRIDE)      // 137280
#define A_ROWSTRIDE 80                // 32*2 bytes + 16 pad   (16B aligned)
#define A_ROWS      (KW * 128)        // [k][o_local]
#define A_BYTES     (A_ROWS * A_ROWSTRIDE)      // 30720
#define SMEM_BYTES  (B_BYTES + 2 * A_BYTES)     // 198720  (< 320KB/WGP)

#if __has_builtin(__builtin_amdgcn_global_load_async_to_lds_b128)
#define HAVE_ASYNC_LDS 1
#else
#define HAVE_ASYNC_LDS 0
#endif

__device__ __forceinline__ void async_copy16(const void* gsrc, void* ldst) {
#if HAVE_ASYNC_LDS
  __builtin_amdgcn_global_load_async_to_lds_b128(
      (as1_v4i)(uintptr_t)gsrc,
      (as3_v4i)(unsigned int)(uintptr_t)ldst, 0, 0);
#else
  *(float4*)ldst = *(const float4*)gsrc;
#endif
}

__device__ __forceinline__ void wait_async_lds() {
#if HAVE_ASYNC_LDS
#if __has_builtin(__builtin_amdgcn_s_wait_asynccnt)
  __builtin_amdgcn_s_wait_asynccnt(0);
#else
  asm volatile("s_wait_asynccnt 0" ::: "memory");
#endif
#endif
}

// --------------------------- weight pre-quant ------------------------------
__global__ void qconv_wquant_kernel(const float* __restrict__ w,
                                    const float* __restrict__ w_scale,
                                    const float* __restrict__ a_scale,
                                    _Float16* __restrict__ wq) {
  int idx = blockIdx.x * blockDim.x + threadIdx.x;
  if (idx >= KW * C_OUT * C_IN) return;
  int k = idx / (C_OUT * C_IN);
  int r = idx - k * (C_OUT * C_IN);
  int o = r / C_IN;
  int c = r - o * C_IN;
  float sw = fabsf(w_scale[o]) + QEPS;
  float sa = fabsf(a_scale[c]) + QEPS;
  float wv = w[(size_t)(o * C_IN + c) * KW + k];
  float q  = rintf(wv / sw);
  q = fminf(fmaxf(q, -128.0f), 127.0f);
  wq[(size_t)(k * C_OUT + o) * C_IN + c] = (_Float16)(q * sw * sa);
}

// ------------------------------ main GEMM ----------------------------------
__global__ __launch_bounds__(256)
void qconv_wmma_kernel(const float* __restrict__ x,
                       const _Float16* __restrict__ wq,
                       const float* __restrict__ bias,
                       const float* __restrict__ a_scale,
                       float* __restrict__ out) {
  extern __shared__ char smem[];
  char* Bs    = smem;                   // [l_local][c] f16 (quantized acts)
  char* Abase = smem + B_BYTES;         // ping-pong A chunks: + p*A_BYTES

  const int l0  = blockIdx.x * 128;
  const int b   = blockIdx.y;
  const int tid = threadIdx.x;

  // ---- Phase 1: quantize & transpose-stage full activation slab (once) ----
  const float* xb = x + (size_t)b * C_IN * LEN;
  for (int idx = tid; idx < C_IN * B_ROWS; idx += 256) {
    int c  = idx / B_ROWS;
    int ll = idx - c * B_ROWS;       // consecutive tid -> consecutive l
    int lg = l0 - 1 + ll;
    _Float16 qh = (_Float16)0.0f;
    if (ll < 130 && lg >= 0 && lg < LEN) {
      float s = fabsf(a_scale[c]) + QEPS;
      float q = rintf(xb[(size_t)c * LEN + lg] / s);
      q = fminf(fmaxf(q, -128.0f), 127.0f);
      qh = (_Float16)q;
    }
    *(_Float16*)(Bs + ll * B_ROWSTRIDE + c * 2) = qh;
  }
  __syncthreads();

  const int lane = tid & 31;
  const int wave = tid >> 5;
  const int half = lane >> 4;     // lanes 16..31 take K 8-15 and 24-31
  const int mrow = lane & 15;
  const int wm   = wave & 3;      // 4 waves along M (C_out): 32 rows each
  const int wn   = wave >> 2;     // 2 waves along N (L):     64 cols each

  // stage one A chunk: wq[k][o0..o0+127][c0..c0+31] -> LDS (16B units)
  auto stage_A = [&](char* dst, int o0, int c0) {
#pragma unroll
    for (int s = 0; s < 6; ++s) {
      int u   = tid + 256 * s;      // 0..1535 16-byte units
      int row = u >> 2;             // 0..383  == k*128 + o_local
      int qd  = u & 3;
      int k   = row >> 7;
      int ol  = row & 127;
      const _Float16* src = wq +
          ((size_t)(k * C_OUT + o0 + ol) * C_IN + c0) + qd * 8;
      async_copy16(src, dst + row * A_ROWSTRIDE + qd * 16);
    }
  };

  for (int o0 = 0; o0 < C_OUT; o0 += 128) {
    v8f acc[2][4];
#pragma unroll
    for (int i = 0; i < 2; ++i)
#pragma unroll
      for (int j = 0; j < 4; ++j)
#pragma unroll
        for (int r = 0; r < 8; ++r) acc[i][j][r] = 0.0f;

    // prologue: buffers free after the barrier; stage chunk 0 into buf 0
    stage_A(Abase, o0, 0);
    wait_async_lds();
    __syncthreads();

    for (int c0 = 0; c0 < C_IN; c0 += 32) {
      const int p = (c0 >> 5) & 1;

      // overlap: kick off next chunk into the other buffer
      if (c0 + 32 < C_IN) {
        stage_A(Abase + (p ^ 1) * A_BYTES, o0, c0 + 32);
        __builtin_prefetch(wq + (size_t)o0 * C_IN + (c0 + 64), 0, 1);
      }

      // ---- 3 taps x (2 M-tiles x 4 N-tiles) WMMA on current buffer ----
      const char* Ac = Abase + p * A_BYTES;
#pragma unroll
      for (int k = 0; k < KW; ++k) {
        v16h afr[2];
#pragma unroll
        for (int i = 0; i < 2; ++i) {
          int row = k * 128 + wm * 32 + i * 16 + mrow;
          const char* base = Ac + row * A_ROWSTRIDE;
          v8h lo = *(const v8h*)(base + half * 16);        // K 0-7  / 8-15
          v8h hi = *(const v8h*)(base + 32 + half * 16);   // K16-23 / 24-31
          afr[i] = __builtin_shufflevector(lo, hi, 0,1,2,3,4,5,6,7,
                                                   8,9,10,11,12,13,14,15);
        }
#pragma unroll
        for (int j = 0; j < 4; ++j) {
          int ll = wn * 64 + j * 16 + mrow + k;            // l_local row
          const char* base = Bs + ll * B_ROWSTRIDE + c0 * 2;
          v8h lo = *(const v8h*)(base + half * 16);        // K 0-7  / 8-15
          v8h hi = *(const v8h*)(base + 32 + half * 16);   // K16-23 / 24-31
          v16h bfr = __builtin_shufflevector(lo, hi, 0,1,2,3,4,5,6,7,
                                                     8,9,10,11,12,13,14,15);
          acc[0][j] = __builtin_amdgcn_wmma_f32_16x16x32_f16(
              false, afr[0], false, bfr, (short)0, acc[0][j], false, false);
          acc[1][j] = __builtin_amdgcn_wmma_f32_16x16x32_f16(
              false, afr[1], false, bfr, (short)0, acc[1][j], false, false);
        }
      }

      // next buffer must be resident; everyone done reading current buffer
      wait_async_lds();
      __syncthreads();
    }

    // ---- epilogue: bias add + store (l-contiguous across lanes 0..15) ----
#pragma unroll
    for (int i = 0; i < 2; ++i) {
#pragma unroll
      for (int r = 0; r < 8; ++r) {
        int og = o0 + wm * 32 + i * 16 + r + 8 * half;     // D: M = r + 8*half
        float bi = bias[og];
        float* orow = out + ((size_t)b * C_OUT + og) * LEN
                          + l0 + wn * 64 + mrow;
#pragma unroll
        for (int j = 0; j < 4; ++j)
          orow[j * 16] = acc[i][j][r] + bi;
      }
    }
  }
}

// ------------------------------- launcher ----------------------------------
extern "C" void kernel_launch(void* const* d_in, const int* in_sizes, int n_in,
                              void* d_out, int out_size, void* d_ws, size_t ws_size,
                              hipStream_t stream) {
  (void)in_sizes; (void)n_in; (void)out_size; (void)ws_size;
  const float* x       = (const float*)d_in[0];
  const float* w       = (const float*)d_in[1];
  const float* bias    = (const float*)d_in[2];
  const float* a_scale = (const float*)d_in[3];
  const float* w_scale = (const float*)d_in[4];
  float*    out = (float*)d_out;
  _Float16* wq  = (_Float16*)d_ws;   // 3*512*512 f16 = 1.5 MB scratch

  (void)hipFuncSetAttribute((const void*)qconv_wmma_kernel,
                            hipFuncAttributeMaxDynamicSharedMemorySize,
                            SMEM_BYTES);

  int nq = KW * C_OUT * C_IN;
  qconv_wquant_kernel<<<(nq + 255) / 256, 256, 0, stream>>>(w, w_scale, a_scale, wq);

  dim3 grid(LEN / 128, BATCH);
  qconv_wmma_kernel<<<grid, 256, SMEM_BYTES, stream>>>(x, wq, bias, a_scale, out);
}
